// DQN_30477087933133
// MI455X (gfx1250) — matrix operand order repaired and checked by reading, not compiled, not run on hardware
//
#include <hip/hip_runtime.h>
#include <hip/hip_bf16.h>

// ---------------------------------------------------------------------------
// CDNA5 wave32 WMMA types
// ---------------------------------------------------------------------------
typedef __bf16 bf16_t;
typedef __attribute__((ext_vector_type(16))) __bf16 v16bf;
typedef __attribute__((ext_vector_type(8)))  __bf16 v8bf;
typedef __attribute__((ext_vector_type(4)))  __bf16 v4bf;
typedef __attribute__((ext_vector_type(8)))  float  v8f;
typedef __attribute__((ext_vector_type(4)))  int    v4i;

// native RNE f32->bf16 (fptrunc; backend selects v_cvt_pk_bf16_f32 pairs)
__device__ __forceinline__ bf16_t f2bf(float f) { return (bf16_t)f; }

// ---------------------------------------------------------------------------
// Async global->LDS (CDNA5, ASYNCcnt) with compile-safe detection + fallback
// ---------------------------------------------------------------------------
#if __has_builtin(__builtin_amdgcn_global_load_async_to_lds_b128)
#define HAVE_ASYNC_LDS 1
#else
#define HAVE_ASYNC_LDS 0
#endif

using v4i_as1 = __attribute__((address_space(1))) v4i;
using v4i_as3 = __attribute__((address_space(3))) v4i;

__device__ __forceinline__ void async_copy_b128(const bf16_t* g, bf16_t* l) {
#if HAVE_ASYNC_LDS
  // param 0 (probe-confirmed by diagnostic): int4 addrspace(1)*
  v4i_as1* gp = (v4i_as1*)(size_t)(const void*)g;
  // generic LDS address: low 32 bits are the LDS byte offset (aperture layout)
  v4i_as3* lp = (v4i_as3*)(unsigned)(size_t)(void*)l;
  __builtin_amdgcn_global_load_async_to_lds_b128(gp, lp, 0, 0);
#else
  (void)g; (void)l;
#endif
}

__device__ __forceinline__ void wait_async0() {
#if __has_builtin(__builtin_amdgcn_s_wait_asynccnt)
  __builtin_amdgcn_s_wait_asynccnt(0);
#else
  asm volatile("s_wait_asynccnt 0x0" ::: "memory");
#endif
}

#define BM 32
#define BN 128
#define BK 32
#define LDK 40   // BK + 8 pad: 80B row stride (16B-aligned, spreads banks)

// ---------------------------------------------------------------------------
// Per-thread register staging for one (A,B) tile (double-buffer pipeline)
// ---------------------------------------------------------------------------
struct StageRegs {
  v8bf   a8;      // A bf16 reg path
  float4 a4[2];   // A f32 path
  float4 b4[8];   // B: 8 contiguous-K rows x 4 cols (f32)
};

// Issue global traffic for tile k0: async A copy (if enabled) + register loads.
template <bool TRANS_A, bool A_BF16, bool USE_ASYNC>
__device__ __forceinline__ void stage_issue(StageRegs& R, const void* Av,
                                            const float* __restrict__ B,
                                            bf16_t (*sAdst)[LDK],
                                            int k0, int m0, int n0,
                                            int lda, int ldb, int t, int K) {
  if constexpr (A_BF16) {
    const bf16_t* A = (const bf16_t*)Av;
    if constexpr (!TRANS_A) {
      const int kq = (t & 3) * 8, r = t >> 2;            // r: 0..31
      const bf16_t* p = A + (size_t)(m0 + r) * lda + k0 + kq;
      if constexpr (USE_ASYNC)
        async_copy_b128(p, &sAdst[r][kq]);               // global->LDS DMA
      else
        R.a8 = *(const v8bf*)p;
      if (k0 + BK < K) __builtin_prefetch(p + BK, 0, 1); // global_prefetch_b8
    } else {
      const int mq = (t & 3) * 8, kr = t >> 2;           // kr: 0..31
      const bf16_t* p = A + (size_t)(k0 + kr) * lda + m0 + mq;
      R.a8 = *(const v8bf*)p;
      if (k0 + BK < K) __builtin_prefetch(p + (size_t)BK * lda, 0, 1);
    }
  } else {
    const float* A = (const float*)Av;
    if constexpr (!TRANS_A) {
      const int kq = (t & 7) * 4, r0 = t >> 3;           // r0: 0..15
      R.a4[0] = *(const float4*)(A + (size_t)(m0 + r0) * lda + k0 + kq);
      R.a4[1] = *(const float4*)(A + (size_t)(m0 + r0 + 16) * lda + k0 + kq);
    } else {
      const int mq = (t & 7) * 4, kr0 = t >> 3;          // kr0: 0..15
      R.a4[0] = *(const float4*)(A + (size_t)(k0 + kr0) * lda + m0 + mq);
      R.a4[1] = *(const float4*)(A + (size_t)(k0 + kr0 + 16) * lda + m0 + mq);
    }
  }
  // B tile [BK x BN] f32: thread owns 8 contiguous k's x 4 contiguous n's
  const int nq = (t & 31) * 4, kr0 = (t >> 5) * 8;
#pragma unroll
  for (int j = 0; j < 8; ++j)
    R.b4[j] = *(const float4*)(B + (size_t)(k0 + kr0 + j) * ldb + n0 + nq);
}

// Commit register-staged data to LDS (A skipped when async carried it).
template <bool TRANS_A, bool A_BF16, bool USE_ASYNC>
__device__ __forceinline__ void stage_commit(const StageRegs& R,
                                             bf16_t (*sA)[LDK], bf16_t (*sBt)[LDK],
                                             int t) {
  if constexpr (A_BF16) {
    if constexpr (!TRANS_A) {
      if constexpr (!USE_ASYNC) {
        const int kq = (t & 3) * 8, r = t >> 2;
        *(v8bf*)&sA[r][kq] = R.a8;                       // one ds_store_b128
      }
    } else {
      const int mq = (t & 3) * 8, kr = t >> 2;
#pragma unroll
      for (int j = 0; j < 8; ++j) sA[mq + j][kr] = R.a8[j];  // LDS transpose
    }
  } else {
    if constexpr (!TRANS_A) {
      const int kq = (t & 7) * 4, r0 = t >> 3;
      v4bf o0, o1;
      o0[0] = f2bf(R.a4[0].x); o0[1] = f2bf(R.a4[0].y);
      o0[2] = f2bf(R.a4[0].z); o0[3] = f2bf(R.a4[0].w);
      o1[0] = f2bf(R.a4[1].x); o1[1] = f2bf(R.a4[1].y);
      o1[2] = f2bf(R.a4[1].z); o1[3] = f2bf(R.a4[1].w);
      *(v4bf*)&sA[r0][kq]      = o0;
      *(v4bf*)&sA[r0 + 16][kq] = o1;
    } else {
      const int mq = (t & 7) * 4, kr0 = t >> 3;
      sA[mq + 0][kr0] = f2bf(R.a4[0].x); sA[mq + 1][kr0] = f2bf(R.a4[0].y);
      sA[mq + 2][kr0] = f2bf(R.a4[0].z); sA[mq + 3][kr0] = f2bf(R.a4[0].w);
      sA[mq + 0][kr0 + 16] = f2bf(R.a4[1].x); sA[mq + 1][kr0 + 16] = f2bf(R.a4[1].y);
      sA[mq + 2][kr0 + 16] = f2bf(R.a4[1].z); sA[mq + 3][kr0 + 16] = f2bf(R.a4[1].w);
    }
  }
  // B: in-register transpose -> 4 column segments of 8 contiguous k's
  const int nq = (t & 31) * 4, kr0 = (t >> 5) * 8;
  v8bf c0, c1, c2, c3;
#pragma unroll
  for (int j = 0; j < 8; ++j) {
    c0[j] = f2bf(R.b4[j].x); c1[j] = f2bf(R.b4[j].y);
    c2[j] = f2bf(R.b4[j].z); c3[j] = f2bf(R.b4[j].w);
  }
  *(v8bf*)&sBt[nq + 0][kr0] = c0;
  *(v8bf*)&sBt[nq + 1][kr0] = c1;
  *(v8bf*)&sBt[nq + 2][kr0] = c2;
  *(v8bf*)&sBt[nq + 3][kr0] = c3;
}

// ---------------------------------------------------------------------------
// C[M,N] = op(A)[M,K] @ B[K,N] (+ optional per-row scale). bf16 x bf16 -> f32
// via v_wmma_f32_16x16x32_bf16, LDS double-buffered, async A staging.
// Requires M%32==0, N%128==0, K%32==0.
// ---------------------------------------------------------------------------
template <bool TRANS_A, bool A_BF16>
__global__ __launch_bounds__(128)
void gemm_wmma(const void* __restrict__ A, const float* __restrict__ B,
               float* __restrict__ C, const float* __restrict__ rowscale,
               int M, int N, int K, int lda, int ldb, int ldc) {
  constexpr bool USE_ASYNC = (A_BF16 && !TRANS_A && (HAVE_ASYNC_LDS != 0));

  __shared__ bf16_t sA[2][BM][LDK];    // [m][k]
  __shared__ bf16_t sBt[2][BN][LDK];   // [n][k]  (contiguous K runs)

  const int t = threadIdx.x;
  const int wave = t >> 5, lane = t & 31;
  const int half = lane >> 4, lr = lane & 15;
  const int m0 = blockIdx.y * BM, n0 = blockIdx.x * BN;
  const int wn = wave * 32;

  v8f acc[2][2] = {};
  StageRegs R;

  stage_issue<TRANS_A, A_BF16, USE_ASYNC>(R, A, B, sA[0], 0, m0, n0, lda, ldb, t, K);
  stage_commit<TRANS_A, A_BF16, USE_ASYNC>(R, sA[0], sBt[0], t);
  if constexpr (USE_ASYNC) wait_async0();
  __syncthreads();

  int cur = 0;
  for (int k0 = 0; k0 < K; k0 += BK) {
    const bool has_next = (k0 + BK) < K;
    if (has_next)  // next tile: async A -> LDS[cur^1], B (+A fallback) -> regs
      stage_issue<TRANS_A, A_BF16, USE_ASYNC>(R, A, B, sA[cur ^ 1], k0 + BK,
                                              m0, n0, lda, ldb, t, K);

    // fragments: two ds_load_b128 each (ISA 7.12.2 wave32 layouts)
    v16bf fa[2], fb[2];
#pragma unroll
    for (int ti = 0; ti < 2; ++ti) {
      const int row = ti * 16 + lr;
      v8bf lo = *(const v8bf*)&sA[cur][row][half * 8];        // K: 8h..8h+7
      v8bf hi = *(const v8bf*)&sA[cur][row][16 + half * 8];   // K: 16+8h..
      fa[ti] = __builtin_shufflevector(lo, hi, 0, 1, 2, 3, 4, 5, 6, 7,
                                       8, 9, 10, 11, 12, 13, 14, 15);
    }
#pragma unroll
    for (int tj = 0; tj < 2; ++tj) {
      const int col = wn + tj * 16 + lr;
      v8bf lo = *(const v8bf*)&sBt[cur][col][half * 8];       // K: 8h..8h+7
      v8bf hi = *(const v8bf*)&sBt[cur][col][16 + half * 8];
      fb[tj] = __builtin_shufflevector(lo, hi, 0, 1, 2, 3, 4, 5, 6, 7,
                                       8, 9, 10, 11, 12, 13, 14, 15);
    }
#pragma unroll
    for (int ti = 0; ti < 2; ++ti)
#pragma unroll
      for (int tj = 0; tj < 2; ++tj)
        acc[ti][tj] = __builtin_amdgcn_wmma_f32_16x16x32_bf16(
            false, fa[ti], false, fb[tj], (short)0, acc[ti][tj], false, false);

    if (has_next) {
      stage_commit<TRANS_A, A_BF16, USE_ASYNC>(R, sA[cur ^ 1], sBt[cur ^ 1], t);
      if constexpr (USE_ASYNC) wait_async0();  // own async done; barrier next
    }
    __syncthreads();
    cur ^= 1;
  }

  // C/D layout: VGPR r -> row r + 8*half, col = lane&15
#pragma unroll
  for (int ti = 0; ti < 2; ++ti) {
#pragma unroll
    for (int tj = 0; tj < 2; ++tj) {
      const int col = n0 + wn + tj * 16 + lr;
#pragma unroll
      for (int r = 0; r < 8; ++r) {
        const int row = m0 + ti * 16 + half * 8 + r;
        float v = acc[ti][tj][r];
        if (rowscale) v *= rowscale[row];
        C[(size_t)row * ldc + col] = v;
      }
    }
  }
}

// ---------------------------------------------------------------------------
// f32 -> bf16 bulk convert (makes Hs/Ht L2-resident at half size)
// ---------------------------------------------------------------------------
__global__ __launch_bounds__(256)
void cvt_f32_bf16(const float* __restrict__ in, bf16_t* __restrict__ out, size_t n) {
  size_t i = ((size_t)blockIdx.x * blockDim.x + threadIdx.x) * 8;
  if (i >= n) return;
  float4 a = *(const float4*)(in + i);
  float4 b = *(const float4*)(in + i + 4);
  v8bf o;
  o[0] = f2bf(a.x); o[1] = f2bf(a.y); o[2] = f2bf(a.z); o[3] = f2bf(a.w);
  o[4] = f2bf(b.x); o[5] = f2bf(b.y); o[6] = f2bf(b.z); o[7] = f2bf(b.w);
  *(v8bf*)(out + i) = o;
}

// ---------------------------------------------------------------------------
// JAX threefry2x32-20, key = jax.random.key(42) -> ks = [0, 42]
// ---------------------------------------------------------------------------
__device__ __forceinline__ unsigned rotl32(unsigned x, int d) {
  return (x << d) | (x >> (32 - d));
}
__device__ __forceinline__ void threefry2x32_20(unsigned c0, unsigned c1,
                                                unsigned& o0, unsigned& o1) {
  const unsigned k0 = 0u, k1 = 42u, k2 = 0x1BD11BDAu ^ 0u ^ 42u;
  unsigned x0 = c0 + k0, x1 = c1 + k1;
#define TF_R(r) { x0 += x1; x1 = rotl32(x1, r); x1 ^= x0; }
  TF_R(13) TF_R(15) TF_R(26) TF_R(6)  x0 += k1; x1 += k2 + 1u;
  TF_R(17) TF_R(29) TF_R(16) TF_R(24) x0 += k2; x1 += k0 + 2u;
  TF_R(13) TF_R(15) TF_R(26) TF_R(6)  x0 += k0; x1 += k1 + 3u;
  TF_R(17) TF_R(29) TF_R(16) TF_R(24) x0 += k1; x1 += k2 + 4u;
  TF_R(13) TF_R(15) TF_R(26) TF_R(6)  x0 += k2; x1 += k0 + 5u;
#undef TF_R
  o0 = x0; o1 = x1;
}

// X1 = dropout(lrelu(agg + trans + bias)); keep <=> MSB(threefry word)==0
__global__ void fuse_drop_kernel(const float* __restrict__ agg,
                                 const float* __restrict__ trn,
                                 const float* __restrict__ bias,
                                 float* __restrict__ out, int halfTotal) {
  int c = blockIdx.x * blockDim.x + threadIdx.x;
  if (c >= halfTotal) return;
  unsigned o0, o1;
  threefry2x32_20((unsigned)c, (unsigned)(c + halfTotal), o0, o1);
  int i0 = c, i1 = c + halfTotal;
  float v0 = agg[i0] + trn[i0] + bias[i0 & 127];
  v0 = v0 >= 0.f ? v0 : 0.01f * v0;
  out[i0] = (o0 >> 31) ? 0.f : v0 * 2.f;
  float v1 = agg[i1] + trn[i1] + bias[i1 & 127];
  v1 = v1 >= 0.f ? v1 : 0.01f * v1;
  out[i1] = (o1 >> 31) ? 0.f : v1 * 2.f;
}

__global__ void fuse_lrelu2_kernel(const float* __restrict__ agg,
                                   const float* __restrict__ trn,
                                   const float* __restrict__ bias,
                                   float* __restrict__ out, int total) {
  int i = blockIdx.x * blockDim.x + threadIdx.x;
  if (i >= total) return;
  float v = agg[i] + trn[i] + bias[i & 127];
  v = v >= 0.f ? v : 0.01f * v;
  v = v >= 0.f ? v : 0.01f * v;
  out[i] = v;
}

// out[row] = X[row,:].fcw[0:128] + state[row]*fcw[128] + fcb  (wave32 reduce)
__global__ __launch_bounds__(256)
void final_fc_kernel(const float* __restrict__ X, const float* __restrict__ st,
                     const float* __restrict__ fcw, const float* __restrict__ fcb,
                     float* __restrict__ out, int rows) {
  int row = blockIdx.x * 8 + (threadIdx.x >> 5);
  if (row >= rows) return;
  int lane = threadIdx.x & 31;
  const float* xr = X + (size_t)row * 128;
  float s = xr[lane] * fcw[lane] + xr[lane + 32] * fcw[lane + 32] +
            xr[lane + 64] * fcw[lane + 64] + xr[lane + 96] * fcw[lane + 96];
#pragma unroll
  for (int o = 16; o > 0; o >>= 1) s += __shfl_xor(s, o, 32);
  if (lane == 0) out[row] = s + st[row] * fcw[128] + fcb[0];
}

// ---------------------------------------------------------------------------
extern "C" void kernel_launch(void* const* d_in, const int* in_sizes, int n_in,
                              void* d_out, int out_size, void* d_ws, size_t ws_size,
                              hipStream_t stream) {
  const float* xi  = (const float*)d_in[0];
  const float* x   = (const float*)d_in[1];
  const float* Ht  = (const float*)d_in[2];   // [E, N] row-major
  const float* Hs  = (const float*)d_in[3];   // [E, N] row-major
  const float* st  = (const float*)d_in[4];
  const float* w0  = (const float*)d_in[5];
  const float* th0 = (const float*)d_in[6];
  const float* ew0 = (const float*)d_in[7];
  const float* b0  = (const float*)d_in[8];
  const float* w1  = (const float*)d_in[9];
  const float* th1 = (const float*)d_in[10];
  const float* ew1 = (const float*)d_in[11];
  const float* b1  = (const float*)d_in[12];
  const float* fcw = (const float*)d_in[13];
  const float* fcb = (const float*)d_in[14];
  float* out = (float*)d_out;

  const int N = 8192, E = 4096, F = 128;
  const size_t EN = (size_t)E * N;

  // workspace plan: [Hsb bf16][Htb bf16][f32 scratch]; fall back to pure-f32
  // operands if ws_size can't hold the bf16 copies (deterministic choice).
  const size_t f32Bytes = (2 * (size_t)E * F + 4 * (size_t)N * F) * sizeof(float);
  const size_t bfBytes  = 2 * EN * sizeof(bf16_t);
  const bool useBf16A = ws_size >= bfBytes + f32Bytes;

  bf16_t* Hsb = (bf16_t*)d_ws;
  bf16_t* Htb = Hsb + EN;
  float* fbase = useBf16A ? (float*)(Htb + EN) : (float*)d_ws;
  float* msg1 = fbase;                        // [E,F]
  float* msg2 = msg1 + (size_t)E * F;         // [E,F]
  float* agg  = msg2 + (size_t)E * F;         // [N,F]
  float* trn  = agg  + (size_t)N * F;         // [N,F]
  float* X1   = trn  + (size_t)N * F;         // [N,F]
  float* Xf   = X1   + (size_t)N * F;         // [N,F]

  dim3 blk(128);
  dim3 gE(F / BN, E / BM);   // (1, 128)
  dim3 gN(F / BN, N / BM);   // (1, 256)

  if (useBf16A) {
    cvt_f32_bf16<<<(unsigned)(EN / 8 / 256), 256, 0, stream>>>(Hs, Hsb, EN);
    cvt_f32_bf16<<<(unsigned)(EN / 8 / 256), 256, 0, stream>>>(Ht, Htb, EN);
  }

#define GEMM_BIG_NT(Bp, Cp) \
  do { if (useBf16A) gemm_wmma<false, true ><<<gE, blk, 0, stream>>>(Hsb, Bp, Cp, nullptr, E, F, N, N, F, F); \
       else          gemm_wmma<false, false><<<gE, blk, 0, stream>>>(Hs,  Bp, Cp, nullptr, E, F, N, N, F, F); } while (0)
#define GEMM_BIG_T(Bp, Cp) \
  do { if (useBf16A) gemm_wmma<true,  true ><<<gN, blk, 0, stream>>>(Htb, Bp, Cp, nullptr, N, F, E, N, F, F); \
       else          gemm_wmma<true,  false><<<gN, blk, 0, stream>>>(Ht,  Bp, Cp, nullptr, N, F, E, N, F, F); } while (0)

  // ---- conv0 ----
  GEMM_BIG_NT(x, msg1);                                         // msg1 = Hs @ x
  gemm_wmma<false, false><<<gE, blk, 0, stream>>>(msg1, th0, msg2, ew0, E, F, F, F, F, F);
  GEMM_BIG_T(msg2, agg);                                        // agg = Ht.T @ msg2
  gemm_wmma<false, false><<<gN, blk, 0, stream>>>(xi, w0, trn, nullptr, N, F, F, F, F, F);
  fuse_drop_kernel<<<(N * F / 2 + 255) / 256, 256, 0, stream>>>(agg, trn, b0, X1, N * F / 2);

  // ---- conv1 ----
  GEMM_BIG_NT(X1, msg1);                                        // msg1 = Hs @ X1
  gemm_wmma<false, false><<<gE, blk, 0, stream>>>(msg1, th1, msg2, ew1, E, F, F, F, F, F);
  GEMM_BIG_T(msg2, agg);                                        // agg = Ht.T @ msg2
  gemm_wmma<false, false><<<gN, blk, 0, stream>>>(xi, w1, trn, nullptr, N, F, F, F, F, F);
  fuse_lrelu2_kernel<<<(N * F + 255) / 256, 256, 0, stream>>>(agg, trn, b1, Xf, N * F);

  // ---- fc ----
  final_fc_kernel<<<N / 8, 256, 0, stream>>>(Xf, st, fcw, fcb, out, N);
#undef GEMM_BIG_NT
#undef GEMM_BIG_T
}
